// CausalSelfAttention_30142080483610
// MI455X (gfx1250) — matrix operand order repaired and checked by reading, not compiled, not run on hardware
//
#include <hip/hip_runtime.h>

#define B_SZ 2
#define T_SZ 2048
#define HID  2048
#define NH   16
#define HD   128
#define M_TOT (B_SZ * T_SZ)   /* 4096 */
#define LPITCH 40             /* LDS row pitch in bf16 elems (80B): conflict-free frag reads */

typedef __attribute__((ext_vector_type(16))) __bf16 bf16x16;
typedef __attribute__((ext_vector_type(8)))  __bf16 bf16x8;
typedef __attribute__((ext_vector_type(8)))  float  v8f;
typedef __attribute__((ext_vector_type(4)))  int    v4i;

static __device__ __forceinline__ __bf16 f2bf(float f) {
  unsigned u = __builtin_bit_cast(unsigned, f);
  unsigned r = u + 0x7FFFu + ((u >> 16) & 1u);   // round-to-nearest-even
  unsigned short s = (unsigned short)(r >> 16);
  return __builtin_bit_cast(__bf16, s);
}
static __device__ __forceinline__ float bf2f(__bf16 b) {
  unsigned short s = __builtin_bit_cast(unsigned short, b);
  return __builtin_bit_cast(float, ((unsigned)s) << 16);
}
static __device__ __forceinline__ bf16x16 cat8(bf16x8 lo, bf16x8 hi) {
  return __builtin_shufflevector(lo, hi, 0,1,2,3,4,5,6,7,8,9,10,11,12,13,14,15);
}
static __device__ __forceinline__ v8f wmma_bf16(bf16x16 a, bf16x16 b, v8f c) {
  return __builtin_amdgcn_wmma_f32_16x16x32_bf16(false, a, false, b, (short)0, c,
                                                 false, false);
}
static __device__ __forceinline__ float half_max(float v) {
#pragma unroll
  for (int m = 1; m < 16; m <<= 1) v = fmaxf(v, __shfl_xor(v, m, 32));
  return v;
}
static __device__ __forceinline__ float half_sum(float v) {
#pragma unroll
  for (int m = 1; m < 16; m <<= 1) v += __shfl_xor(v, m, 32);
  return v;
}

// ---- async global->LDS copy (ASYNCcnt-tracked on gfx1250), with fallback -----
#if defined(__has_builtin)
#if __has_builtin(__builtin_amdgcn_global_load_async_to_lds_b128)
#define HAVE_ASYNC_LDS 1
#endif
#if __has_builtin(__builtin_amdgcn_s_wait_asynccnt)
#define HAVE_WAIT_ASYNC 1
#endif
#endif

static __device__ __forceinline__ void cp16(__bf16* ldst, const __bf16* gsrc) {
#ifdef HAVE_ASYNC_LDS
  __builtin_amdgcn_global_load_async_to_lds_b128(
      (__attribute__((address_space(1))) v4i*)gsrc,
      (__attribute__((address_space(3))) v4i*)ldst, 0, 0);
#else
  *(bf16x8*)ldst = *(const bf16x8*)gsrc;
#endif
}
static __device__ __forceinline__ void wait_async_keep3() {
#ifdef HAVE_WAIT_ASYNC
  __builtin_amdgcn_s_wait_asynccnt(3);   // previous stage's 3 copies done; newest 3 in flight
#endif
}
static __device__ __forceinline__ void wait_async_all() {
#ifdef HAVE_WAIT_ASYNC
  __builtin_amdgcn_s_wait_asynccnt(0);
#endif
}

// ---------------------------------------------------------------- cast x -> bf16
__global__ void cast_f32_bf16(const float* __restrict__ in, __bf16* __restrict__ out, int n) {
  int i = blockIdx.x * blockDim.x + threadIdx.x;
  if (i < n) out[i] = f2bf(in[i]);
}

// ------------------------------------------- W[K,N] f32  ->  Wt[N,K] bf16
__global__ void transpose_cast_w(const float* __restrict__ W, __bf16* __restrict__ Wt,
                                 int K, int N) {
  __shared__ float tile[32][33];
  int tx = threadIdx.x, ty = threadIdx.y;            // 32 x 8
  int n0 = blockIdx.x * 32, k0 = blockIdx.y * 32;
#pragma unroll
  for (int j = 0; j < 32; j += 8)
    tile[ty + j][tx] = W[(size_t)(k0 + ty + j) * N + n0 + tx];
  __syncthreads();
#pragma unroll
  for (int j = 0; j < 32; j += 8)
    Wt[(size_t)(n0 + ty + j) * K + k0 + tx] = f2bf(tile[tx][ty + j]);
}

// -------------------------------- vp [B,T,H,D] bf16 -> vt [B,H,D,T] bf16
__global__ void transpose_v(const __bf16* __restrict__ vp, __bf16* __restrict__ vt) {
  __shared__ __bf16 tile[32][33];
  int bh = blockIdx.z, b = bh / NH, h = bh % NH;
  int tx = threadIdx.x, ty = threadIdx.y;            // 32 x 8
  int d0 = blockIdx.x * 32, t0 = blockIdx.y * 32;
#pragma unroll
  for (int j = 0; j < 32; j += 8)
    tile[ty + j][tx] = vp[(size_t)(b * T_SZ + t0 + ty + j) * HID + h * HD + d0 + tx];
  __syncthreads();
#pragma unroll
  for (int j = 0; j < 32; j += 8)
    vt[((size_t)(b * NH + h) * HD + d0 + ty + j) * T_SZ + t0 + tx] = tile[tx][ty + j];
}

// ---------------------------------------------------- RoPE in-place on q and k
__global__ void rope_inplace(__bf16* __restrict__ q, __bf16* __restrict__ k) {
  int idx = blockIdx.x * blockDim.x + threadIdx.x;   // B*T*H*(D/2) = 2^22
  int i = idx & 63;
  int h = (idx >> 6) & (NH - 1);
  int t = (idx >> 10) & (T_SZ - 1);
  int b = idx >> 21;
  if (b >= B_SZ) return;
  size_t base = (size_t)(b * T_SZ + t) * HID + h * HD;
  float inv = __expf(-(float)(2 * i) * (9.210340371976184f / (float)HD)); // 10000^(-2i/D)
  float ang = (float)t * inv;
  float s, c;
  __sincosf(ang, &s, &c);
  float q0 = bf2f(q[base + i]), q1 = bf2f(q[base + i + 64]);
  float k0 = bf2f(k[base + i]), k1 = bf2f(k[base + i + 64]);
  q[base + i]      = f2bf(q0 * c - q1 * s);
  q[base + i + 64] = f2bf(q1 * c + q0 * s);
  k[base + i]      = f2bf(k0 * c - k1 * s);
  k[base + i + 64] = f2bf(k1 * c + k0 * s);
}

// ---- C[M,N] = A[M,K]bf16 @ Bt[N,K]bf16^T ------------------------------------
// 256 thr = 8 waves (4x2), block tile 128x64, K-stage 32, double-buffered LDS
// staged with async global->LDS copies (3 b128 ops per thread per stage).
template <bool OUT_BF16>
__global__ void __launch_bounds__(256)
gemm_bf16(const __bf16* __restrict__ A, const __bf16* __restrict__ Bt,
          void* __restrict__ Cout, int M, int N, int K) {
  __shared__ __bf16 sA[2][128 * LPITCH];   // 2 x 10,240 B
  __shared__ __bf16 sB[2][64 * LPITCH];    // 2 x  5,120 B
  const int tid  = threadIdx.x;
  const int lane = tid & 31, wave = tid >> 5;
  const int grp  = lane >> 4, lm = lane & 15;
  const int wm   = wave >> 1, wn = wave & 1;
  const int blockM = blockIdx.y * 128;
  const int blockN = blockIdx.x * 64;

  const int crow  = tid >> 2;        // 0..63
  const int cpart = tid & 3;         // 16B quarter of a 64B row segment

  auto issue_stage = [&](int buf, int k0) {
    // A tile: 128 rows x 32 elems; rows crow and crow+64
    cp16(&sA[buf][crow * LPITCH + cpart * 8],
         A + (size_t)(blockM + crow) * K + k0 + cpart * 8);
    cp16(&sA[buf][(crow + 64) * LPITCH + cpart * 8],
         A + (size_t)(blockM + crow + 64) * K + k0 + cpart * 8);
    // B tile: 64 rows x 32 elems
    cp16(&sB[buf][crow * LPITCH + cpart * 8],
         Bt + (size_t)(blockN + crow) * K + k0 + cpart * 8);
  };

  v8f acc[2][2] = {};
  issue_stage(0, 0);
  const int nstage = K / 32;
  for (int i = 0; i < nstage; ++i) {
    const int cur = i & 1;
    if (i + 1 < nstage) {
      issue_stage(cur ^ 1, (i + 1) * 32);
      wait_async_keep3();              // stage i copies complete (in-order)
    } else {
      wait_async_all();
    }
    __syncthreads();                   // all waves' copies visible

    bf16x16 afrag[2], bfrag[2];
#pragma unroll
    for (int tm = 0; tm < 2; ++tm) {
      const __bf16* la = &sA[cur][(wm * 32 + tm * 16 + lm) * LPITCH + grp * 8];
      afrag[tm] = cat8(*(const bf16x8*)la, *(const bf16x8*)(la + 16));
    }
#pragma unroll
    for (int tn = 0; tn < 2; ++tn) {
      const __bf16* lb = &sB[cur][(wn * 32 + tn * 16 + lm) * LPITCH + grp * 8];
      bfrag[tn] = cat8(*(const bf16x8*)lb, *(const bf16x8*)(lb + 16));
    }
#pragma unroll
    for (int tm = 0; tm < 2; ++tm)
#pragma unroll
      for (int tn = 0; tn < 2; ++tn)
        acc[tm][tn] = wmma_bf16(afrag[tm], bfrag[tn], acc[tm][tn]);

    __syncthreads();                   // reads done before buffer is re-filled
  }

#pragma unroll
  for (int tm = 0; tm < 2; ++tm)
#pragma unroll
    for (int tn = 0; tn < 2; ++tn)
#pragma unroll
      for (int r = 0; r < 8; ++r) {
        int row = blockM + wm * 32 + tm * 16 + grp * 8 + r;
        int col = blockN + wn * 32 + tn * 16 + lm;
        if (OUT_BF16)
          ((__bf16*)Cout)[(size_t)row * N + col] = f2bf(acc[tm][tn][r]);
        else
          ((float*)Cout)[(size_t)row * N + col] = acc[tm][tn][r];
      }
}

// ---------------------------------------------- flash attention, 1 wave / 16 q rows
__global__ void __launch_bounds__(256)
flash_attn(const __bf16* __restrict__ q, const __bf16* __restrict__ k,
           const __bf16* __restrict__ vt, __bf16* __restrict__ y) {
  __shared__ __bf16 pbuf[8 * 512];                   // 16x32 bf16 per wave
  const int lane = threadIdx.x & 31, wv = threadIdx.x >> 5;
  const int grp = lane >> 4, lm = lane & 15;
  const int wid = blockIdx.x * 8 + wv;               // 4096 waves total
  const int qt = wid & 127, bh = wid >> 7;
  const int h = bh & (NH - 1), b = bh >> 4;
  const int q0 = qt * 16;
  __bf16* lds = pbuf + wv * 512;

  bf16x16 qa[4];
#pragma unroll
  for (int kc = 0; kc < 4; ++kc) {
    const __bf16* ap = q + (size_t)(b * T_SZ + q0 + lm) * HID + h * HD + kc * 32 + grp * 8;
    qa[kc] = cat8(*(const bf16x8*)ap, *(const bf16x8*)(ap + 16));
  }

  v8f o[8] = {};
  float mrow[8], lrow[8];
#pragma unroll
  for (int r = 0; r < 8; ++r) { mrow[r] = -1.0e30f; lrow[r] = 0.0f; }

  const float scale = 0.08838834764831845f;          // 1/sqrt(128)
  const __bf16* vbase = vt + (size_t)(b * NH + h) * HD * T_SZ;
  const int nkb = (q0 + 16 + 31) >> 5;

  for (int kb = 0; kb < nkb; ++kb) {
    v8f s[2];
#pragma unroll
    for (int kt = 0; kt < 2; ++kt) {
      v8f acc = {};
      int n0 = kb * 32 + kt * 16;
#pragma unroll
      for (int kc = 0; kc < 4; ++kc) {
        const __bf16* kp = k + (size_t)(b * T_SZ + n0 + lm) * HID + h * HD + kc * 32 + grp * 8;
        acc = wmma_bf16(qa[kc], cat8(*(const bf16x8*)kp, *(const bf16x8*)(kp + 16)), acc);
      }
      s[kt] = acc;
    }
#pragma unroll
    for (int r = 0; r < 8; ++r) {
      int row = q0 + grp * 8 + r;
      float v0 = s[0][r] * scale; if (kb * 32 + lm > row)      v0 = -1.0e30f;
      float v1 = s[1][r] * scale; if (kb * 32 + 16 + lm > row) v1 = -1.0e30f;
      float mn = half_max(fmaxf(v0, v1));
      mn = fmaxf(mn, mrow[r]);
      float alpha = __expf(mrow[r] - mn);
      mrow[r] = mn;
      float p0 = __expf(v0 - mn), p1 = __expf(v1 - mn);
      lrow[r] = lrow[r] * alpha + half_sum(p0 + p1);
#pragma unroll
      for (int dt = 0; dt < 8; ++dt) o[dt][r] *= alpha;
      lds[(grp * 8 + r) * 32 + lm]      = f2bf(p0);
      lds[(grp * 8 + r) * 32 + 16 + lm] = f2bf(p1);
    }
    // same-wave DS ops are in-order (ISA 7.3): no barrier needed
    bf16x16 pa = cat8(*(const bf16x8*)(lds + lm * 32 + grp * 8),
                      *(const bf16x8*)(lds + lm * 32 + 16 + grp * 8));
#pragma unroll
    for (int dt = 0; dt < 8; ++dt) {
      const __bf16* vp = vbase + (size_t)(dt * 16 + lm) * T_SZ + kb * 32 + grp * 8;
      o[dt] = wmma_bf16(pa, cat8(*(const bf16x8*)vp, *(const bf16x8*)(vp + 16)), o[dt]);
    }
  }
#pragma unroll
  for (int r = 0; r < 8; ++r) {
    float inv = 1.0f / lrow[r];
    size_t rbase = (size_t)(b * T_SZ + q0 + grp * 8 + r) * HID + h * HD;
#pragma unroll
    for (int dt = 0; dt < 8; ++dt)
      y[rbase + dt * 16 + lm] = f2bf(o[dt][r] * inv);
  }
}

extern "C" void kernel_launch(void* const* d_in, const int* in_sizes, int n_in,
                              void* d_out, int out_size, void* d_ws, size_t ws_size,
                              hipStream_t stream) {
  const float* x  = (const float*)d_in[0];
  const float* Wq = (const float*)d_in[1];
  const float* Wk = (const float*)d_in[2];
  const float* Wv = (const float*)d_in[3];
  const float* Wo = (const float*)d_in[4];

  const size_t ACT = (size_t)M_TOT * HID * sizeof(__bf16);   // 16 MB
  const size_t WTS = (size_t)HID * HID * sizeof(__bf16);     // 8 MB
  char* p = (char*)d_ws;
  __bf16* xb  = (__bf16*)p; p += ACT;
  __bf16* wqt = (__bf16*)p; p += WTS;
  __bf16* wkt = (__bf16*)p; p += WTS;
  __bf16* wvt = (__bf16*)p; p += WTS;
  __bf16* wot = (__bf16*)p; p += WTS;
  __bf16* qp  = (__bf16*)p; p += ACT;
  __bf16* kp  = (__bf16*)p; p += ACT;
  __bf16* vp  = (__bf16*)p; p += ACT;
  __bf16* vtb = (__bf16*)p; p += ACT;
  __bf16* yb  = (__bf16*)p; p += ACT;

  const int n_elem = M_TOT * HID;
  cast_f32_bf16<<<(n_elem + 255) / 256, 256, 0, stream>>>(x, xb, n_elem);

  dim3 tb(32, 8), tg(HID / 32, HID / 32);
  transpose_cast_w<<<tg, tb, 0, stream>>>(Wq, wqt, HID, HID);
  transpose_cast_w<<<tg, tb, 0, stream>>>(Wk, wkt, HID, HID);
  transpose_cast_w<<<tg, tb, 0, stream>>>(Wv, wvt, HID, HID);
  transpose_cast_w<<<tg, tb, 0, stream>>>(Wo, wot, HID, HID);

  dim3 gg(HID / 64, M_TOT / 128);
  gemm_bf16<true><<<gg, 256, 0, stream>>>(xb, wqt, qp, M_TOT, HID, HID);
  gemm_bf16<true><<<gg, 256, 0, stream>>>(xb, wkt, kp, M_TOT, HID, HID);
  gemm_bf16<true><<<gg, 256, 0, stream>>>(xb, wvt, vp, M_TOT, HID, HID);

  const int n_rope = B_SZ * T_SZ * NH * (HD / 2);
  rope_inplace<<<n_rope / 256, 256, 0, stream>>>(qp, kp);

  dim3 vg(HD / 32, T_SZ / 32, B_SZ * NH);
  transpose_v<<<vg, tb, 0, stream>>>(vp, vtb);

  flash_attn<<<(B_SZ * NH * (T_SZ / 16)) / 8, 256, 0, stream>>>(qp, kp, vtb, yb);

  gemm_bf16<false><<<gg, 256, 0, stream>>>(yb, wot, d_out, M_TOT, HID, HID);
}
